// AttGRU_40089224741047
// MI455X (gfx1250) — matrix operand order repaired and checked by reading, not compiled, not run on hardware
//
#include <hip/hip_runtime.h>
#include <math.h>

#define BDIM 32
#define SLEN 512
#define DDIM 1024

typedef __attribute__((ext_vector_type(16))) __bf16 v16bf;
typedef __attribute__((ext_vector_type(8)))  __bf16 v8bf;
typedef __attribute__((ext_vector_type(8)))  float  v8f;

static constexpr size_t NQ     = (size_t)BDIM * DDIM;          // 32768 floats
static constexpr size_t NGATES = (size_t)SLEN * BDIM;          // 16384 floats
static constexpr size_t NMEM   = (size_t)BDIM * SLEN * DDIM;   // 16.7M elems
static constexpr size_t NW3    = (size_t)3 * DDIM * DDIM;      // 3.1M elems
static constexpr size_t NGI    = (size_t)SLEN * 3 * BDIM * DDIM; // 50.3M elems

__device__ __forceinline__ unsigned short f32_to_bf16_rne(float f) {
    unsigned int u = __float_as_uint(f);
    unsigned int r = u + 0x7FFFu + ((u >> 16) & 1u);
    return (unsigned short)(r >> 16);
}

// ---------------------------------------------------------------- casts
__global__ void k_f32_to_bf16(const float* __restrict__ src,
                              unsigned short* __restrict__ dst, size_t n) {
    size_t i = (size_t)blockIdx.x * blockDim.x + threadIdx.x;
    size_t stride = (size_t)gridDim.x * blockDim.x;
    for (; i < n; i += stride) dst[i] = f32_to_bf16_rne(src[i]);
}

// ---------------------------------------------------------------- q = query @ Wq.T + bq
__global__ void k_q(const float* __restrict__ query, const float* __restrict__ Wq,
                    const float* __restrict__ bq, float* __restrict__ q) {
    __shared__ float qs[DDIM];
    int b = blockIdx.x;
    for (int d = threadIdx.x; d < DDIM; d += blockDim.x)
        qs[d] = query[(size_t)b * DDIM + d];
    __syncthreads();
    int e = blockIdx.y * blockDim.x + threadIdx.x;
    const float* wrow = Wq + (size_t)e * DDIM;
    float acc = 0.f;
    for (int d = 0; d < DDIM; ++d) acc += qs[d] * wrow[d];
    q[(size_t)b * DDIM + e] = acc + bq[e];
}

// ---------------------------------------------------------------- p = q @ Wu
__global__ void k_p(const float* __restrict__ q, const float* __restrict__ Wu,
                    float* __restrict__ p) {
    __shared__ float qs[DDIM];
    int b = blockIdx.x;
    for (int d = threadIdx.x; d < DDIM; d += blockDim.x)
        qs[d] = q[(size_t)b * DDIM + d];
    __syncthreads();
    int d0 = blockIdx.y * blockDim.x + threadIdx.x;
    float acc = 0.f;
    for (int e = 0; e < DDIM; ++e) acc += qs[e] * Wu[(size_t)e * DDIM + d0];
    p[(size_t)b * DDIM + d0] = acc;
}

// ---------------------------------------------------------------- attention logits + softmax -> gates[s*B+b]
__global__ void k_attn(const float* __restrict__ q, const float* __restrict__ bu,
                       const float* __restrict__ p, const float* __restrict__ memory,
                       const unsigned char* __restrict__ mask, float* __restrict__ gates) {
    __shared__ float ps[DDIM];
    __shared__ float logits[SLEN];
    __shared__ float red[256];
    int b = blockIdx.x, tid = threadIdx.x;
    for (int d = tid; d < DDIM; d += 256) ps[d] = p[(size_t)b * DDIM + d];
    float partial = 0.f;
    for (int d = tid; d < DDIM; d += 256) partial += q[(size_t)b * DDIM + d] * bu[d];
    red[tid] = partial; __syncthreads();
    for (int s = 128; s; s >>= 1) { if (tid < s) red[tid] += red[tid + s]; __syncthreads(); }
    float qb = red[0];
    __syncthreads();
    int wave = tid >> 5, lane = tid & 31;
    for (int s = wave; s < SLEN; s += 8) {
        const float* mrow = memory + ((size_t)b * SLEN + s) * DDIM;
        float acc = 0.f;
        for (int j = lane; j < DDIM; j += 32) acc += mrow[j] * ps[j];
        for (int off = 16; off; off >>= 1) acc += __shfl_xor(acc, off, 32);
        if (lane == 0) {
            float v = acc + qb;
            if (mask[(size_t)b * SLEN + s]) v = -INFINITY;
            logits[s] = v;
        }
    }
    __syncthreads();
    float mx = -INFINITY;
    for (int s = tid; s < SLEN; s += 256) mx = fmaxf(mx, logits[s]);
    red[tid] = mx; __syncthreads();
    for (int s = 128; s; s >>= 1) { if (tid < s) red[tid] = fmaxf(red[tid], red[tid + s]); __syncthreads(); }
    mx = red[0]; __syncthreads();
    float sm = 0.f;
    for (int s = tid; s < SLEN; s += 256) sm += __expf(logits[s] - mx);
    red[tid] = sm; __syncthreads();
    for (int s = 128; s; s >>= 1) { if (tid < s) red[tid] += red[tid + s]; __syncthreads(); }
    float inv = 1.f / red[0];
    __syncthreads();
    for (int s = tid; s < SLEN; s += 256)
        gates[s * BDIM + b] = __expf(logits[s] - mx) * inv;
}

// ---------------------------------------------------------------- WMMA fragment helpers
__device__ __forceinline__ v16bf load_a_frag(const unsigned short* row, int ka, int kb) {
    v8bf lo = *(const v8bf*)(row + ka + 2 * kb);
    v8bf hi = *(const v8bf*)(row + ka + 2 * kb + 8);
    v16bf a;
#pragma unroll
    for (int j = 0; j < 8; ++j) { a[j] = lo[j]; a[8 + j] = hi[j]; }
    return a;
}

// ---------------------------------------------------------------- GI = mem @ Wi.T + bi (both dirs), WMMA bf16
// Output rows r = s*32 + b  (s = scan step; dir 1 reads memory reversed in s)
__global__ void k_gi(const unsigned short* __restrict__ memA,
                     const unsigned short* __restrict__ WiF,
                     const unsigned short* __restrict__ WiB,
                     const float* __restrict__ biF, const float* __restrict__ biB,
                     float* __restrict__ GIF, float* __restrict__ GIB) {
    const int dir = blockIdx.z;
    const unsigned short* Wi = dir ? WiB : WiF;
    const float* bi = dir ? biB : biF;
    float* GI = dir ? GIB : GIF;
    const int lane = threadIdx.x & 31, wave = threadIdx.x >> 5;
    const int rbase = blockIdx.x * 32;
    const int cbase = blockIdx.y * 256 + wave * 32;
    const int m = lane & 15, n = lane & 15;
    const int kb = (lane >> 4) * 8, kb2 = (lane >> 4) * 16;

    int r0 = rbase + m, r1 = rbase + 16 + m;
    int s0 = r0 >> 5, b0 = r0 & 31, s1 = r1 >> 5, b1 = r1 & 31;
    int sm0 = dir ? (SLEN - 1 - s0) : s0;
    int sm1 = dir ? (SLEN - 1 - s1) : s1;
    const unsigned short* arow0 = memA + ((size_t)b0 * SLEN + sm0) * DDIM;
    const unsigned short* arow1 = memA + ((size_t)b1 * SLEN + sm1) * DDIM;
    const unsigned short* brow0 = Wi + (size_t)(cbase + n) * DDIM;
    const unsigned short* brow1 = Wi + (size_t)(cbase + 16 + n) * DDIM;

    v8f acc00 = {}, acc01 = {}, acc10 = {}, acc11 = {};
    for (int kt = 0; kt < DDIM / 32; ++kt) {
        int ka = kt * 32;
        v16bf a0 = load_a_frag(arow0, ka, kb);
        v16bf a1 = load_a_frag(arow1, ka, kb);
        v16bf bb0 = *(const v16bf*)(brow0 + ka + kb2);
        v16bf bb1 = *(const v16bf*)(brow1 + ka + kb2);
        acc00 = __builtin_amdgcn_wmma_f32_16x16x32_bf16(false, a0, false, bb0, (short)0, acc00, false, false);
        acc01 = __builtin_amdgcn_wmma_f32_16x16x32_bf16(false, a0, false, bb1, (short)0, acc01, false, false);
        acc10 = __builtin_amdgcn_wmma_f32_16x16x32_bf16(false, a1, false, bb0, (short)0, acc10, false, false);
        acc11 = __builtin_amdgcn_wmma_f32_16x16x32_bf16(false, a1, false, bb1, (short)0, acc11, false, false);
    }
    const int mb = (lane < 16) ? 0 : 8;
    const int c0 = cbase + n, c1 = cbase + 16 + n;
    const float bi0 = bi[c0], bi1 = bi[c1];
#pragma unroll
    for (int i = 0; i < 8; ++i) {
        int rr = rbase + mb + i;
        int ss = rr >> 5, bb = rr & 31;
        GI[(((size_t)ss * 3 + (c0 >> 10)) * 32 + bb) * DDIM + (c0 & 1023)] = acc00[i] + bi0;
        GI[(((size_t)ss * 3 + (c1 >> 10)) * 32 + bb) * DDIM + (c1 & 1023)] = acc01[i] + bi1;
        int rr1 = rbase + 16 + mb + i;
        int ss1 = rr1 >> 5, bb1i = rr1 & 31;
        GI[(((size_t)ss1 * 3 + (c0 >> 10)) * 32 + bb1i) * DDIM + (c0 & 1023)] = acc10[i] + bi0;
        GI[(((size_t)ss1 * 3 + (c1 >> 10)) * 32 + bb1i) * DDIM + (c1 & 1023)] = acc11[i] + bi1;
    }
}

// ---------------------------------------------------------------- grid-wide barrier (monotonic counter)
__device__ __forceinline__ void grid_barrier(unsigned int* bar, unsigned int target) {
    __builtin_amdgcn_fence(__ATOMIC_RELEASE, "agent");
    __syncthreads();
    if (threadIdx.x == 0) {
        atomicAdd(bar, 1u);
        while (__hip_atomic_load(bar, __ATOMIC_RELAXED, __HIP_MEMORY_SCOPE_AGENT) < target)
            __builtin_amdgcn_s_sleep(2);
    }
    __syncthreads();
    __builtin_amdgcn_fence(__ATOMIC_ACQUIRE, "agent");
}

// ---------------------------------------------------------------- persistent recurrent scan
// 32 WGs: dir = blk>>4, each WG owns e-slice [slice*64, slice*64+64)
__global__ void k_rec(const float* __restrict__ GIF, const float* __restrict__ GIB,
                      const unsigned short* __restrict__ WhF, const unsigned short* __restrict__ WhB,
                      const float* __restrict__ bhF, const float* __restrict__ bhB,
                      const float* __restrict__ gates, const float* __restrict__ hidden,
                      float* __restrict__ hf32, unsigned short* __restrict__ hb16,
                      unsigned int* bar, float* __restrict__ out) {
    const int dir = blockIdx.x >> 4;
    const int slice = blockIdx.x & 15;
    const int e0 = slice * 64;
    const float* GI = dir ? GIB : GIF;
    const unsigned short* Wh = dir ? WhB : WhF;
    const float* bh = dir ? bhB : bhF;
    float* h = hf32 + (size_t)dir * BDIM * DDIM;
    const int tid = threadIdx.x, lane = tid & 31, wave = tid >> 5;
    __shared__ float ghs[32][192];

    // init this WG's h slice (double-buffered bf16 copy in buffer 0)
    for (int idx = tid; idx < 32 * 64; idx += 256) {
        int b = idx >> 6, el = idx & 63, e = e0 + el;
        float v = hidden[((size_t)dir * BDIM + b) * DDIM + e];
        h[(size_t)b * DDIM + e] = v;
        hb16[((size_t)dir) * BDIM * DDIM + (size_t)b * DDIM + e] = f32_to_bf16_rne(v);
    }
    unsigned int barIdx = 1;
    grid_barrier(bar, barIdx * 32);

    const int m = lane & 15, n = lane & 15;
    const int kb = (lane >> 4) * 8, kb2 = (lane >> 4) * 16;
    const int mb = (lane < 16) ? 0 : 8;

    for (int s = 0; s < SLEN; ++s) {
        const int cur = s & 1, nxt = cur ^ 1;
        const unsigned short* hA = hb16 + ((size_t)cur * 2 + dir) * BDIM * DDIM;
        // gh = h @ Wh.T on this WG's 3x64 column slice: 24 16x16 tiles, 3 per wave
        for (int t = wave; t < 24; t += 8) {
            const int mrow = t / 12, ncol = t % 12;
            const int b = mrow * 16 + m;
            const int cl = ncol * 16 + n;              // 0..191
            const int k = cl / 64, e = e0 + (cl % 64);
            const unsigned short* arow = hA + (size_t)b * DDIM;
            const unsigned short* brow = Wh + ((size_t)k * DDIM + e) * DDIM;
            v8f acc = {};
            for (int kt = 0; kt < DDIM / 32; ++kt) {
                int ka = kt * 32;
                v16bf a = load_a_frag(arow, ka, kb);
                v16bf bb = *(const v16bf*)(brow + ka + kb2);
                if (kt + 1 < DDIM / 32)
                    __builtin_prefetch((const void*)(brow + ka + 32 + kb2), 0, 3);
                acc = __builtin_amdgcn_wmma_f32_16x16x32_bf16(false, a, false, bb, (short)0, acc, false, false);
            }
#pragma unroll
            for (int i = 0; i < 8; ++i) ghs[mrow * 16 + mb + i][cl] = acc[i];
        }
        __syncthreads();
        // elementwise GRU cell update on owned slice
        for (int idx = tid; idx < 32 * 64; idx += 256) {
            int b = idx >> 6, el = idx & 63, e = e0 + el;
            size_t gibase = (((size_t)s * 3) * 32 + b) * DDIM + e;
            float gi0 = GI[gibase];
            float gi1 = GI[gibase + (size_t)32 * DDIM];
            float gi2 = GI[gibase + (size_t)64 * DDIM];
            float gh0 = ghs[b][el]        + bh[e];
            float gh1 = ghs[b][64 + el]   + bh[DDIM + e];
            float gh2 = ghs[b][128 + el]  + bh[2 * DDIM + e];
            int smem = dir ? (SLEN - 1 - s) : s;
            float g  = gates[smem * BDIM + b];
            float r  = 1.f / (1.f + __expf(-(gi0 + gh0)));
            float z  = 1.f / (1.f + __expf(-(gi1 + gh1)));
            float gg = 1.f / (1.f + __expf(-(g + z)));
            float nn = tanhf(gi2 + r * gh2);
            float hold = h[(size_t)b * DDIM + e];
            float hnew = gg * nn + (1.f - gg) * hold;
            h[(size_t)b * DDIM + e] = hnew;
            hb16[((size_t)nxt * 2 + dir) * BDIM * DDIM + (size_t)b * DDIM + e] = f32_to_bf16_rne(hnew);
        }
        ++barIdx;
        grid_barrier(bar, barIdx * 32);
    }
    // final h -> output (concat hf, hb)
    for (int idx = tid; idx < 32 * 64; idx += 256) {
        int b = idx >> 6, el = idx & 63, e = e0 + el;
        out[((size_t)dir * BDIM + b) * DDIM + e] = h[(size_t)b * DDIM + e];
    }
}

// ----------------------------------------------------------------
extern "C" void kernel_launch(void* const* d_in, const int* in_sizes, int n_in,
                              void* d_out, int out_size, void* d_ws, size_t ws_size,
                              hipStream_t stream) {
    (void)in_sizes; (void)n_in; (void)out_size; (void)ws_size;
    const float* query  = (const float*)d_in[0];
    const float* memory = (const float*)d_in[1];
    const float* hidden = (const float*)d_in[2];
    const unsigned char* mask = (const unsigned char*)d_in[3];
    const float* Wq  = (const float*)d_in[4];
    const float* bq  = (const float*)d_in[5];
    const float* Wu  = (const float*)d_in[6];
    const float* bu  = (const float*)d_in[7];
    const float* WiF = (const float*)d_in[8];
    const float* biF = (const float*)d_in[9];
    const float* WhF = (const float*)d_in[10];
    const float* bhF = (const float*)d_in[11];
    const float* WiB = (const float*)d_in[12];
    const float* biB = (const float*)d_in[13];
    const float* WhB = (const float*)d_in[14];
    const float* bhB = (const float*)d_in[15];

    char* w = (char*)d_ws;
    auto take = [&](size_t bytes) -> char* {
        char* r = w; w += (bytes + 255) & ~(size_t)255; return r;
    };
    float*          q    = (float*)take(NQ * 4);
    float*          p    = (float*)take(NQ * 4);
    float*          gate = (float*)take(NGATES * 4);
    unsigned short* memA = (unsigned short*)take(NMEM * 2);
    unsigned short* wiF  = (unsigned short*)take(NW3 * 2);
    unsigned short* wiB  = (unsigned short*)take(NW3 * 2);
    unsigned short* whF  = (unsigned short*)take(NW3 * 2);
    unsigned short* whB  = (unsigned short*)take(NW3 * 2);
    float*          GIF  = (float*)take(NGI * 4);
    float*          GIB  = (float*)take(NGI * 4);
    float*          hf32 = (float*)take((size_t)2 * NQ * 4);
    unsigned short* hb16 = (unsigned short*)take((size_t)4 * NQ * 2);
    unsigned int*   bar  = (unsigned int*)take(256);

    hipMemsetAsync(bar, 0, sizeof(unsigned int), stream);

    k_f32_to_bf16<<<4096, 256, 0, stream>>>(memory, memA, NMEM);
    k_f32_to_bf16<<<2048, 256, 0, stream>>>(WiF, wiF, NW3);
    k_f32_to_bf16<<<2048, 256, 0, stream>>>(WiB, wiB, NW3);
    k_f32_to_bf16<<<2048, 256, 0, stream>>>(WhF, whF, NW3);
    k_f32_to_bf16<<<2048, 256, 0, stream>>>(WhB, whB, NW3);

    k_q<<<dim3(32, 4), 256, 0, stream>>>(query, Wq, bq, q);
    k_p<<<dim3(32, 4), 256, 0, stream>>>(q, Wu, p);
    k_attn<<<32, 256, 0, stream>>>(q, bu, p, memory, mask, gate);

    k_gi<<<dim3(SLEN * BDIM / 32, 3 * DDIM / 256, 2), 256, 0, stream>>>(
        memA, wiF, wiB, biF, biB, GIF, GIB);

    k_rec<<<32, 256, 0, stream>>>(GIF, GIB, whF, whB, bhF, bhB, gate, hidden,
                                  hf32, hb16, bar, (float*)d_out);
}
